// SelfAttention_78812649881770
// MI455X (gfx1250) — matrix operand order repaired and checked by reading, not compiled, not run on hardware
//
#include <hip/hip_runtime.h>
#include <hip/hip_bf16.h>
#include <stdint.h>

typedef __bf16 bf16_t;
typedef __bf16 v16bf __attribute__((ext_vector_type(16)));
typedef __bf16 v8bf  __attribute__((ext_vector_type(8)));
typedef float  v8f   __attribute__((ext_vector_type(8)));

#define DIM_IN 128
#define DIM_K  64
#define DIM_V  128
#define SEQ    4096
#define BATCH  8

__device__ __forceinline__ v16bf cat8(v8bf lo, v8bf hi) {
  return __builtin_shufflevector(lo, hi, 0,1,2,3,4,5,6,7,8,9,10,11,12,13,14,15);
}

// Async Global -> LDS copy of 16 bytes (per lane), tracked by ASYNCcnt.
// lds_addr: wave-relative LDS byte address (low 32 bits of flat shared ptr,
// per flat-LDS aperture truncation, ISA 10.2). gaddr: 64-bit global address.
__device__ __forceinline__ void async_copy_b128(uint32_t lds_addr, uint64_t gaddr) {
  asm volatile("global_load_async_to_lds_b128 %0, %1, off"
               :: "v"(lds_addr), "v"(gaddr)
               : "memory");
}
__device__ __forceinline__ void wait_asynccnt0() {
  asm volatile("s_wait_asynccnt 0x0" ::: "memory");
}

// ---------------------------------------------------------------------------
// Kernel 1: QKV projection. One block per (b, n). x column staged in LDS,
// 256 threads = 64 q dims + 64 k dims + 128 v dims. fp32 math, bf16 output.
// ---------------------------------------------------------------------------
__global__ __launch_bounds__(256)
void qkv_proj_kernel(const float* __restrict__ x,
                     const float* __restrict__ Wq,
                     const float* __restrict__ Wk,
                     const float* __restrict__ Wv,
                     bf16_t* __restrict__ Q,
                     bf16_t* __restrict__ K,
                     bf16_t* __restrict__ V) {
  const int bn = blockIdx.x;            // b * SEQ + n
  const int b  = bn >> 12;              // /4096
  const int n  = bn & (SEQ - 1);
  __shared__ float y[DIM_IN];
  const int t = threadIdx.x;
  if (t < DIM_IN) y[t] = x[((size_t)b * DIM_IN + t) * SEQ + n];
  __syncthreads();

  const float* w;
  bf16_t* dst;
  if (t < 64)       { w = Wq + t * DIM_IN;        dst = Q + (size_t)bn * DIM_K + t; }
  else if (t < 128) { w = Wk + (t - 64) * DIM_IN; dst = K + (size_t)bn * DIM_K + (t - 64); }
  else              { w = Wv + (t - 128) * DIM_IN; dst = V + (size_t)bn * DIM_V + (t - 128); }

  float acc = 0.f;
#pragma unroll
  for (int d = 0; d < DIM_IN; ++d) acc += y[d] * w[d];
  *dst = (bf16_t)acc;
}

// ---------------------------------------------------------------------------
// Kernel 2: fused flash attention + output transpose + residual add.
// Block = 128 threads (4 waves). Wave handles 16 query rows; KV tiled by 64.
// V tile staged with async global->LDS DMA (ASYNCcnt); K tile transposed
// through VGPRs into LDS.
// ---------------------------------------------------------------------------
__global__ __launch_bounds__(128)
void attn_kernel(const bf16_t* __restrict__ Q,
                 const bf16_t* __restrict__ K,
                 const bf16_t* __restrict__ V,
                 const float*  __restrict__ x,
                 float* __restrict__ out) {
  __shared__ __align__(16) bf16_t Kt[64 * 64];        // [d][j] : K tile transposed
  __shared__ __align__(16) bf16_t Vt[64 * DIM_V];     // [j][v] : V tile row-major
  __shared__ __align__(16) bf16_t Pb[4][16 * 64];     // per-wave P tile [16][64]

  const int tid  = threadIdx.x;
  const int wave = tid >> 5;
  const int lane = tid & 31;
  const int half = lane >> 4;
  const int lr   = lane & 15;
  const int b    = blockIdx.y;
  const int m0   = blockIdx.x * 64 + wave * 16;

  const bf16_t* Qb = Q + ((size_t)b * SEQ) * DIM_K;
  const bf16_t* Kb = K + ((size_t)b * SEQ) * DIM_K;
  const bf16_t* Vb = V + ((size_t)b * SEQ) * DIM_V;

  // --- Q A-fragments (16x32 each, K-dim 64 -> 2 frags), kept in VGPRs ---
  v16bf aq[2];
#pragma unroll
  for (int kb = 0; kb < 2; ++kb) {
    const int row = m0 + lr;
    const int off = kb * 32 + half * 8;
    v8bf lo = *reinterpret_cast<const v8bf*>(Qb + (size_t)row * DIM_K + off);
    v8bf hi = *reinterpret_cast<const v8bf*>(Qb + (size_t)row * DIM_K + off + 16);
    aq[kb] = cat8(lo, hi);
  }

  v8f o[8];
  const v8f vzero = {0.f, 0.f, 0.f, 0.f, 0.f, 0.f, 0.f, 0.f};
#pragma unroll
  for (int jo = 0; jo < 8; ++jo) o[jo] = vzero;
  float mrow[8], lrow[8];
#pragma unroll
  for (int r = 0; r < 8; ++r) { mrow[r] = -3.0e38f; lrow[r] = 0.f; }

  // LDS byte address of V tile for async DMA (wave-relative offset)
  const uint32_t vt_lds = (uint32_t)(uintptr_t)(&Vt[0]);

  for (int n0 = 0; n0 < SEQ; n0 += 64) {
    __syncthreads();   // protect LDS reuse from previous iteration readers

    {
      const int j = tid >> 1;
      // --- stage V tile row-major via async global->LDS (no VGPR round-trip)
      const int vbase = (tid & 1) * 64;
      const bf16_t* vsrc = Vb + (size_t)(n0 + j) * DIM_V + vbase;
      const uint32_t vdst = vt_lds + (uint32_t)(j * DIM_V + vbase) * 2u;
#pragma unroll
      for (int c = 0; c < 8; ++c)
        async_copy_b128(vdst + c * 16u, (uint64_t)(uintptr_t)(vsrc + c * 8));

      // --- stage K tile transposed: thread -> row j, 32 dims ---
      const int dbase = (tid & 1) * 32;
      const bf16_t* src = Kb + (size_t)(n0 + j) * DIM_K + dbase;
#pragma unroll
      for (int c = 0; c < 4; ++c) {
        v8bf kv = *reinterpret_cast<const v8bf*>(src + c * 8);
#pragma unroll
        for (int i = 0; i < 8; ++i)
          Kt[(dbase + c * 8 + i) * 64 + j] = kv[i];
      }
    }
    wait_asynccnt0();   // our async V stores are in LDS
    __syncthreads();    // everyone's stores visible

    // --- S = Q * K^T, 16x64 in 4 j-tiles of 16 ---
    v8f s[4];
#pragma unroll
    for (int jt = 0; jt < 4; ++jt) {
      v8f acc = vzero;
#pragma unroll
      for (int kb = 0; kb < 2; ++kb) {
        v16bf bk;
#pragma unroll
        for (int e = 0; e < 16; ++e) {
          const int kk = (e < 8) ? (half * 8 + e) : (16 + half * 8 + (e - 8));
          bk[e] = Kt[(kb * 32 + kk) * 64 + jt * 16 + lr];
        }
        acc = __builtin_amdgcn_wmma_f32_16x16x32_bf16(
            false, aq[kb], false, bk, (short)0, acc, false, false);
      }
      s[jt] = acc * 0.125f;   // 1/sqrt(64)
    }

    // --- online softmax: lane holds rows r + 8*half, column lr ---
#pragma unroll
    for (int r = 0; r < 8; ++r) {
      float rmax = fmaxf(fmaxf(s[0][r], s[1][r]), fmaxf(s[2][r], s[3][r]));
#pragma unroll
      for (int msk = 1; msk < 16; msk <<= 1)
        rmax = fmaxf(rmax, __shfl_xor(rmax, msk, 16));
      const float mnew  = fmaxf(mrow[r], rmax);
      const float alpha = __expf(mrow[r] - mnew);
      mrow[r] = mnew;
      float psum = 0.f;
#pragma unroll
      for (int jt = 0; jt < 4; ++jt) {
        const float p = __expf(s[jt][r] - mnew);
        s[jt][r] = p;
        psum += p;
      }
      lrow[r] = lrow[r] * alpha + psum;   // per-lane partial sum (4 cols)
#pragma unroll
      for (int jo = 0; jo < 8; ++jo) o[jo][r] *= alpha;
    }

    // --- write P (bf16) to per-wave LDS tile (D-layout -> row-major) ---
#pragma unroll
    for (int jt = 0; jt < 4; ++jt)
#pragma unroll
      for (int r = 0; r < 8; ++r)
        Pb[wave][(r + 8 * half) * 64 + jt * 16 + lr] = (bf16_t)s[jt][r];

    // same-wave DS ops are in-order (ISA 7.3): reload as A-fragments
    // --- O += P * V ---
#pragma unroll
    for (int kb = 0; kb < 2; ++kb) {
      const int off = kb * 32 + half * 8;
      v8bf plo = *reinterpret_cast<const v8bf*>(&Pb[wave][lr * 64 + off]);
      v8bf phi = *reinterpret_cast<const v8bf*>(&Pb[wave][lr * 64 + off + 16]);
      v16bf pa = cat8(plo, phi);
#pragma unroll
      for (int jo = 0; jo < 8; ++jo) {
        v16bf bv;
#pragma unroll
        for (int e = 0; e < 16; ++e) {
          const int kk = (e < 8) ? (half * 8 + e) : (16 + half * 8 + (e - 8));
          bv[e] = Vt[(kb * 32 + kk) * DIM_V + jo * 16 + lr];
        }
        o[jo] = __builtin_amdgcn_wmma_f32_16x16x32_bf16(
            false, pa, false, bv, (short)0, o[jo], false, false);
      }
    }
  }

  // --- epilogue: normalize, transpose to [b][v][n], add residual x ---
#pragma unroll
  for (int r = 0; r < 8; ++r) {
    float L = lrow[r];
#pragma unroll
    for (int msk = 1; msk < 16; msk <<= 1)
      L += __shfl_xor(L, msk, 16);
    const float rl = 1.0f / L;
    const int n = m0 + r + 8 * half;
#pragma unroll
    for (int jo = 0; jo < 8; ++jo) {
      const int c = jo * 16 + lr;
      const size_t idx = ((size_t)b * DIM_V + c) * SEQ + n;
      out[idx] = o[jo][r] * rl + x[idx];
    }
  }
}

// ---------------------------------------------------------------------------
extern "C" void kernel_launch(void* const* d_in, const int* in_sizes, int n_in,
                              void* d_out, int out_size, void* d_ws, size_t ws_size,
                              hipStream_t stream) {
  const float* x  = (const float*)d_in[0];
  const float* Wq = (const float*)d_in[1];
  const float* Wk = (const float*)d_in[2];
  const float* Wv = (const float*)d_in[3];
  float* out = (float*)d_out;

  char* ws = (char*)d_ws;
  const size_t qk_bytes = (size_t)BATCH * SEQ * DIM_K * sizeof(bf16_t);  // 4 MB
  bf16_t* Q = (bf16_t*)(ws);
  bf16_t* K = (bf16_t*)(ws + qk_bytes);
  bf16_t* V = (bf16_t*)(ws + 2 * qk_bytes);  // 8 MB

  qkv_proj_kernel<<<BATCH * SEQ, 256, 0, stream>>>(x, Wq, Wk, Wv, Q, K, V);
  attn_kernel<<<dim3(SEQ / 64, BATCH), 128, 0, stream>>>(Q, K, V, x, out);
  (void)in_sizes; (void)n_in; (void)out_size; (void)ws_size;
}
